// MultiHeadAttention_26079041422144
// MI455X (gfx1250) — matrix operand order repaired and checked by reading, not compiled, not run on hardware
//
#include <hip/hip_runtime.h>
#include <cstdint>

// ---------------- problem constants ----------------
#define D_MODEL 1024
#define NHEADS  16
#define HDIM    64
#define SEQ     2048
#define BATCH   2
#define M_ROWS  (BATCH * SEQ)   // 4096

#define AS1 __attribute__((address_space(1)))
#define AS3 __attribute__((address_space(3)))

#if defined(__AMDGCN__) && __has_builtin(__builtin_amdgcn_global_load_async_to_lds_b128)
#define HAVE_ASYNC 1
#else
#define HAVE_ASYNC 0
#endif

// ---------------- vector types ----------------
typedef _Float16 h16v16 __attribute__((ext_vector_type(16)));
typedef _Float16 h16v8  __attribute__((ext_vector_type(8)));
typedef float    f32v8  __attribute__((ext_vector_type(8)));
typedef float    f32v4  __attribute__((ext_vector_type(4)));
// GCC-style vector to exactly match the async builtin's parameter type:
//   'int __vector(4) __device__ *'  (from the round-2 diagnostic)
typedef int v4i_gcc __attribute__((vector_size(4 * sizeof(int))));

static __device__ __forceinline__ f32v8 wmma_f16(h16v16 a, h16v16 b, f32v8 c) {
    // v_wmma_f32_16x16x32_f16  (D = A(16x32) * B(32x16) + C)
    return __builtin_amdgcn_wmma_f32_16x16x32_f16(false, a, false, b, (short)0, c, false, false);
}

// A-fragment (16x32 f16) from f16 row-major [*, ld]:
// lane l: row = l%16 ; K = {0..15} (l<16) or {16..31}
static __device__ __forceinline__ h16v16 load_a_f16(const _Float16* __restrict__ src, int ld,
                                                    int rowBase, int kBase, int lane) {
    const int row  = rowBase + (lane & 15);
    const int koff = (lane < 16) ? 0 : 16;
    const _Float16* p = src + (size_t)row * ld + kBase;
    h16v8 lo = *(const h16v8*)(p + koff);
    h16v8 hi = *(const h16v8*)(p + 8 + koff);
    h16v16 a;
#pragma unroll
    for (int i = 0; i < 8; ++i) { a[i] = lo[i]; a[8 + i] = hi[i]; }
    return a;
}

// B-fragment (32x16 f16) from f16 torch-layout weight W[N,K]  (B(k,n) = W[n][k]):
// lane l: col n = l%16 ; K = kBase + (l<16?0:16) + i -> 16 contiguous f16 (32B aligned)
static __device__ __forceinline__ h16v16 load_b_f16(const _Float16* __restrict__ w, int ld,
                                                    int nBase, int kBase, int lane) {
    const int n  = nBase + (lane & 15);
    const int k0 = kBase + ((lane < 16) ? 0 : 16);
    return *(const h16v16*)(w + (size_t)n * ld + k0);
}

// ============================================================
// Kernel 0: f32 -> f16 conversion (8 elements / thread, exact grids)
// ============================================================
__global__ __launch_bounds__(256) void cvt_f32_to_f16_kernel(const float* __restrict__ src,
                                                             _Float16* __restrict__ dst, int n) {
    const int i = (blockIdx.x * 256 + threadIdx.x) * 8;
    if (i >= n) return;
    f32v4 a = *(const f32v4*)(src + i);
    f32v4 b = *(const f32v4*)(src + i + 4);
    h16v8 h;
#pragma unroll
    for (int j = 0; j < 4; ++j) { h[j] = (_Float16)a[j]; h[4 + j] = (_Float16)b[j]; }
    *(h16v8*)(dst + i) = h;
}

// ============================================================
// Kernel 1: fused QKV projection  qkv = x @ W_qkv^T + b  (all-f16 operands)
//   M=4096, N=3072, K=1024; wave tile 32x64 (8 WMMA / k-step), WG tile 64x256
//   scatter into Q/K/V [B,H,S,64] (f16)
// ============================================================
__global__ __launch_bounds__(256) void qkv_proj_kernel(
    const _Float16* __restrict__ xh, const _Float16* __restrict__ wh,
    const float* __restrict__ bias,
    _Float16* __restrict__ Q, _Float16* __restrict__ K, _Float16* __restrict__ V) {

    const int lane  = threadIdx.x & 31;
    const int wave  = threadIdx.x >> 5;
    const int mBase = blockIdx.x * 64 + (wave >> 2) * 32;
    const int nBase = blockIdx.y * 256 + (wave & 3) * 64;

    f32v8 acc[2][4] = {};
    for (int k = 0; k < D_MODEL; k += 32) {
        h16v16 a0 = load_a_f16(xh, D_MODEL, mBase, k, lane);
        h16v16 a1 = load_a_f16(xh, D_MODEL, mBase + 16, k, lane);
        h16v16 b[4];
#pragma unroll
        for (int t = 0; t < 4; ++t) b[t] = load_b_f16(wh, D_MODEL, nBase + t * 16, k, lane);
        __builtin_prefetch(xh + (size_t)(mBase + (lane & 15)) * D_MODEL + k + 64, 0, 0);
        __builtin_prefetch(wh + (size_t)(nBase + (lane & 15)) * D_MODEL + k + 64, 0, 0);
#pragma unroll
        for (int t = 0; t < 4; ++t) {
            acc[0][t] = wmma_f16(a0, b[t], acc[0][t]);
            acc[1][t] = wmma_f16(a1, b[t], acc[1][t]);
        }
    }

    const int col0   = lane & 15;
    const int rowAdd = (lane < 16) ? 0 : 8;
#pragma unroll
    for (int t = 0; t < 4; ++t) {
        const int e   = nBase + t * 16 + col0;          // column in [0,3072)
        const float bv = bias[e];
        const int h   = e / 192;                        // head
        const int r   = e - h * 192;                    // 0..63 Q | 64..127 K | 128..191 V
        _Float16* buf = (r < 64) ? Q : ((r < 128) ? K : V);
        const int rr  = (r < 64) ? r : ((r < 128) ? (r - 64) : (r - 128));
#pragma unroll
        for (int mt = 0; mt < 2; ++mt) {
#pragma unroll
            for (int j = 0; j < 8; ++j) {
                const int gm = mBase + mt * 16 + j + rowAdd;   // flat (b*SEQ + s)
                const int b  = gm >> 11;
                const int s  = gm & (SEQ - 1);
                buf[(((size_t)b * NHEADS + h) * SEQ + s) * HDIM + rr] =
                    (_Float16)(acc[mt][t][j] + bv);
            }
        }
    }
}

// ============================================================
// Kernel 2: flash attention per (b,h).
//   WG = 128 threads (4 waves); wave owns 16 query rows; WG owns 64.
//   Key blocks of 32, online softmax; K tile staged via ASYNC load-to-LDS.
// ============================================================
__global__ __launch_bounds__(128) void flash_attn_kernel(
    const _Float16* __restrict__ Q, const _Float16* __restrict__ K,
    const _Float16* __restrict__ V, _Float16* __restrict__ ctx) {

    __shared__ __align__(32) _Float16 ldsK[32 * HDIM];   // [key][hd]
    __shared__ __align__(32) _Float16 ldsVt[HDIM * 32];  // [hd][key] (transposed)
    __shared__ __align__(32) _Float16 ldsP[4][16 * 32];  // per-wave P tile [row][key]

    const int lane  = threadIdx.x & 31;
    const int wave  = threadIdx.x >> 5;
    const int bh    = blockIdx.y;                      // b*NHEADS + h
    const int qBase = blockIdx.x * 64 + wave * 16;     // query row within sequence

    const _Float16* Qp = Q + (size_t)bh * SEQ * HDIM;
    const _Float16* Kp = K + (size_t)bh * SEQ * HDIM;
    const _Float16* Vp = V + (size_t)bh * SEQ * HDIM;

    // Q A-fragments for hd chunks [0,32) and [32,64), resident for all key blocks
    h16v16 qa[2];
    {
        const int row  = qBase + (lane & 15);
        const int koff = (lane < 16) ? 0 : 16;
        const _Float16* p = Qp + (size_t)row * HDIM;
#pragma unroll
        for (int c = 0; c < 2; ++c) {
            h16v8 lo = *(const h16v8*)(p + c * 32 + koff);
            h16v8 hi = *(const h16v8*)(p + c * 32 + 8 + koff);
#pragma unroll
            for (int i = 0; i < 8; ++i) { qa[c][i] = lo[i]; qa[c][8 + i] = hi[i]; }
        }
    }

    f32v8 acc[4] = {};
    f32v8 m_row, l_row;
#pragma unroll
    for (int j = 0; j < 8; ++j) { m_row[j] = -1e30f; l_row[j] = 0.0f; }
    const float scale = 0.125f;   // 1/sqrt(64)

    const int sKey = threadIdx.x >> 2;         // 0..31 (staging role)
    const int sHd0 = (threadIdx.x & 3) * 16;

    for (int kb = 0; kb < SEQ; kb += 32) {
        __syncthreads();   // previous block's LDS reads complete before overwrite
        // -------- stage K block [32x64] (async DMA to LDS) --------
        {
            const _Float16* kp = Kp + (size_t)(kb + sKey) * HDIM + sHd0;
#if HAVE_ASYNC
            AS1 v4i_gcc* g = (AS1 v4i_gcc*)(uintptr_t)kp;
            AS3 v4i_gcc* l = (AS3 v4i_gcc*)(unsigned)(uintptr_t)&ldsK[sKey * HDIM + sHd0];
            __builtin_amdgcn_global_load_async_to_lds_b128(g, l, 0, 0);
            __builtin_amdgcn_global_load_async_to_lds_b128(g, l, 16, 0);
#else
            *(h16v8*)&ldsK[sKey * HDIM + sHd0]     = *(const h16v8*)kp;
            *(h16v8*)&ldsK[sKey * HDIM + sHd0 + 8] = *(const h16v8*)(kp + 8);
#endif
        }
        // -------- stage V block transposed [64x32] --------
        {
            const _Float16* vp = Vp + (size_t)(kb + sKey) * HDIM + sHd0;
            h16v8 v0 = *(const h16v8*)vp;
            h16v8 v1 = *(const h16v8*)(vp + 8);
#pragma unroll
            for (int i = 0; i < 8; ++i) {
                ldsVt[(sHd0 + i) * 32 + sKey]     = v0[i];
                ldsVt[(sHd0 + 8 + i) * 32 + sKey] = v1[i];
            }
        }
#if HAVE_ASYNC
        asm volatile("s_wait_asynccnt 0" ::: "memory");
#endif
        __syncthreads();

        // -------- scores S = Q * K^T : two 16x16 C tiles (keys 0-15, 16-31) --------
        f32v8 c0 = {}, c1 = {};
        {
            const int keyCol = lane & 15;
            const int kk     = (lane < 16) ? 0 : 16;
#pragma unroll
            for (int c = 0; c < 2; ++c) {   // hd chunks [0,32), [32,64)
                h16v16 b0 = *(const h16v16*)&ldsK[keyCol * HDIM + c * 32 + kk];
                h16v16 b1 = *(const h16v16*)&ldsK[(16 + keyCol) * HDIM + c * 32 + kk];
                c0 = wmma_f16(qa[c], b0, c0);
                c1 = wmma_f16(qa[c], b1, c1);
            }
        }

        // -------- online softmax (stats in C-layout: index j = row j / j+8) --------
        f32v8 rm;
#pragma unroll
        for (int j = 0; j < 8; ++j) rm[j] = fmaxf(c0[j], c1[j]) * scale;
#pragma unroll
        for (int off = 1; off < 16; off <<= 1) {
#pragma unroll
            for (int j = 0; j < 8; ++j) rm[j] = fmaxf(rm[j], __shfl_xor(rm[j], off, 32));
        }
        f32v8 p0, p1, rs, alpha;
#pragma unroll
        for (int j = 0; j < 8; ++j) {
            const float mn = fmaxf(m_row[j], rm[j]);
            alpha[j] = __expf(m_row[j] - mn);
            m_row[j] = mn;
            const float a0 = __expf(c0[j] * scale - mn);
            const float a1 = __expf(c1[j] * scale - mn);
            p0[j] = a0; p1[j] = a1;
            rs[j] = a0 + a1;
        }
#pragma unroll
        for (int off = 1; off < 16; off <<= 1) {
#pragma unroll
            for (int j = 0; j < 8; ++j) rs[j] += __shfl_xor(rs[j], off, 32);
        }
#pragma unroll
        for (int j = 0; j < 8; ++j) l_row[j] = l_row[j] * alpha[j] + rs[j];
#pragma unroll
        for (int t = 0; t < 4; ++t)
#pragma unroll
            for (int j = 0; j < 8; ++j) acc[t][j] *= alpha[j];

        // -------- P: C-layout -> A-layout via per-wave LDS buffer --------
        {
            _Float16* pbuf = ldsP[wave];
            const int rAdd = (lane < 16) ? 0 : 8;
            const int col  = lane & 15;
#pragma unroll
            for (int j = 0; j < 8; ++j) {
                pbuf[(j + rAdd) * 32 + col]      = (_Float16)p0[j];
                pbuf[(j + rAdd) * 32 + 16 + col] = (_Float16)p1[j];
            }
        }
        asm volatile("s_wait_dscnt 0" ::: "memory");   // same-wave LDS RAW

        h16v16 pa;
        {
            const int row  = lane & 15;
            const int koff = (lane < 16) ? 0 : 16;
            const _Float16* p = ldsP[wave] + row * 32;
            h16v8 lo = *(const h16v8*)(p + koff);
            h16v8 hi = *(const h16v8*)(p + 8 + koff);
#pragma unroll
            for (int i = 0; i < 8; ++i) { pa[i] = lo[i]; pa[8 + i] = hi[i]; }
        }

        // -------- O += P(16x32) * V(32x64) --------
        {
            const int hdCol = lane & 15;
            const int kk    = (lane < 16) ? 0 : 16;
#pragma unroll
            for (int t = 0; t < 4; ++t) {
                h16v16 vb = *(const h16v16*)&ldsVt[(t * 16 + hdCol) * 32 + kk];
                acc[t] = wmma_f16(pa, vb, acc[t]);
            }
        }
    }

    // -------- epilogue: normalize and store ctx[b, s, h*64 + hd] (f16) --------
    {
        const int b    = bh >> 4;
        const int h    = bh & 15;
        const int col0 = lane & 15;
        const int rAdd = (lane < 16) ? 0 : 8;
#pragma unroll
        for (int j = 0; j < 8; ++j) {
            const float inv = 1.0f / l_row[j];
            const int s = qBase + j + rAdd;
            _Float16* dst = ctx + ((size_t)(b * SEQ + s)) * D_MODEL + h * HDIM;
#pragma unroll
            for (int t = 0; t < 4; ++t)
                dst[t * 16 + col0] = (_Float16)(acc[t][j] * inv);
        }
    }
}

// ============================================================
// Kernel 3: output projection  out = ctx @ W_out^T + b (fp32 out)
//   M=4096, N=1024, K=1024; all-f16 operands, wave tile 32x64
// ============================================================
__global__ __launch_bounds__(256) void out_proj_kernel(
    const _Float16* __restrict__ ctx, const _Float16* __restrict__ wh,
    const float* __restrict__ bias, float* __restrict__ out) {

    const int lane  = threadIdx.x & 31;
    const int wave  = threadIdx.x >> 5;
    const int mBase = blockIdx.x * 64 + (wave >> 2) * 32;
    const int nBase = blockIdx.y * 256 + (wave & 3) * 64;

    f32v8 acc[2][4] = {};
    for (int k = 0; k < D_MODEL; k += 32) {
        h16v16 a0 = load_a_f16(ctx, D_MODEL, mBase, k, lane);
        h16v16 a1 = load_a_f16(ctx, D_MODEL, mBase + 16, k, lane);
        h16v16 b[4];
#pragma unroll
        for (int t = 0; t < 4; ++t) b[t] = load_b_f16(wh, D_MODEL, nBase + t * 16, k, lane);
#pragma unroll
        for (int t = 0; t < 4; ++t) {
            acc[0][t] = wmma_f16(a0, b[t], acc[0][t]);
            acc[1][t] = wmma_f16(a1, b[t], acc[1][t]);
        }
    }

    const int col0   = lane & 15;
    const int rowAdd = (lane < 16) ? 0 : 8;
#pragma unroll
    for (int t = 0; t < 4; ++t) {
        const int gn = nBase + t * 16 + col0;
        const float bv = bias[gn];
#pragma unroll
        for (int mt = 0; mt < 2; ++mt) {
#pragma unroll
            for (int j = 0; j < 8; ++j) {
                const int gm = mBase + mt * 16 + j + rowAdd;
                out[(size_t)gm * D_MODEL + gn] = acc[mt][t][j] + bv;
            }
        }
    }
}

// ============================================================
extern "C" void kernel_launch(void* const* d_in, const int* in_sizes, int n_in,
                              void* d_out, int out_size, void* d_ws, size_t ws_size,
                              hipStream_t stream) {
    (void)in_sizes; (void)n_in; (void)out_size; (void)ws_size;
    const float* x     = (const float*)d_in[0];
    const float* w_qkv = (const float*)d_in[1];
    const float* b_qkv = (const float*)d_in[2];
    const float* w_out = (const float*)d_in[3];
    const float* b_out = (const float*)d_in[4];
    float* out = (float*)d_out;

    // ---- workspace layout (all f16) ----
    const size_t xElems   = (size_t)M_ROWS * D_MODEL;        // 4 Mi
    const size_t wqkvEl   = (size_t)3 * D_MODEL * D_MODEL;   // 3 Mi
    const size_t woutEl   = (size_t)D_MODEL * D_MODEL;       // 1 Mi
    const size_t qkvElems = (size_t)BATCH * NHEADS * SEQ * HDIM; // 4 Mi each

    _Float16* xh    = (_Float16*)d_ws;
    _Float16* wqkvh = xh + xElems;
    _Float16* wouth = wqkvh + wqkvEl;
    _Float16* Q     = wouth + woutEl;
    _Float16* K     = Q + qkvElems;
    _Float16* V     = K + qkvElems;
    _Float16* ctx   = V + qkvElems;   // total ~48 MiB of d_ws

    // 0) one-shot f32 -> f16 conversions (removes cvt chains from GEMM hot loops)
    cvt_f32_to_f16_kernel<<<(int)(xElems / 2048), 256, 0, stream>>>(x, xh, (int)xElems);
    cvt_f32_to_f16_kernel<<<(int)(wqkvEl / 2048), 256, 0, stream>>>(w_qkv, wqkvh, (int)wqkvEl);
    cvt_f32_to_f16_kernel<<<(int)(woutEl / 2048), 256, 0, stream>>>(w_out, wouth, (int)woutEl);

    // 1) QKV projection: M=4096 (64 tiles), N=3072 (12 tiles)
    qkv_proj_kernel<<<dim3(64, 12), 256, 0, stream>>>(xh, wqkvh, b_qkv, Q, K, V);
    // 2) flash attention: 32 query blocks x 32 (b,h) pairs
    flash_attn_kernel<<<dim3(SEQ / 64, BATCH * NHEADS), 128, 0, stream>>>(Q, K, V, ctx);
    // 3) output projection: M=4096 (64), N=1024 (4)
    out_proj_kernel<<<dim3(64, 4), 256, 0, stream>>>(ctx, wouth, b_out, out);
}
